// MeshCNNClassifier_6940667150713
// MI455X (gfx1250) — matrix operand order repaired and checked by reading, not compiled, not run on hardware
//
#include <hip/hip_runtime.h>
#include <hip/hip_bf16.h>

typedef _Float16 h2  __attribute__((ext_vector_type(2)));
typedef _Float16 h8  __attribute__((ext_vector_type(8)));
typedef _Float16 v16h __attribute__((ext_vector_type(16)));
typedef float    v8f __attribute__((ext_vector_type(8)));

#define LN_EPS 1e-5f
#define ROWH 328   // padded feats row length (halfs): 656B -> good LDS bank spread, 16B aligned

static __device__ __forceinline__ h2 h2abs(h2 v) {
    unsigned u = __builtin_bit_cast(unsigned, v) & 0x7fff7fffu;
    return __builtin_bit_cast(h2, u);
}

// ---------------------------------------------------------------------------
// Pack w_rest (3 x 320 x 64 f32, row-major [K][N]) into f16 WMMA B-fragments.
// Read side (mesh kernel): lane reads 16 contiguous halfs at
//   wp[ ((ks*4 + ntile)*32 + lane)*16 .. +15 ]
// which must equal W[ks*32 + 16*(h>=8) + 8*(lane>=16) + (h&7)][ntile*16 + (lane&15)].
// ---------------------------------------------------------------------------
__global__ __launch_bounds__(256)
void pack_w_kernel(const float* __restrict__ w_rest, _Float16* __restrict__ wp) {
    int t = blockIdx.x * 256 + threadIdx.x;
    if (t >= 3 * 20480) return;
    int layer = t / 20480;
    int rem   = t % 20480;
    int frag  = rem >> 9;          // 0..39  (= ks*4 + ntile)
    int lane  = (rem >> 4) & 31;
    int h     = rem & 15;
    int ks = frag >> 2;
    int nt = frag & 3;
    int K = ks * 32 + ((h & 8) ? 16 : 0) + ((lane & 16) ? 8 : 0) + (h & 7);
    int N = nt * 16 + (lane & 15);
    wp[t] = (_Float16)w_rest[layer * 20480 + K * 64 + N];
}

// ---------------------------------------------------------------------------
// Layer 0: C_IN=11 -> K=55 (WMMA-unfriendly, ~5% of FLOPs): scalar f32.
// One thread per edge. Weights are wave-uniform -> scalar-cached loads.
// Writes f32 residual stream + f16 gather copy.
// ---------------------------------------------------------------------------
__global__ __launch_bounds__(256)
void layer0_kernel(const float* __restrict__ x, const int4* __restrict__ nbr,
                   const float* __restrict__ w0, const float* __restrict__ b0,
                   const float* __restrict__ g, const float* __restrict__ be,
                   float* __restrict__ xres, _Float16* __restrict__ xout, int E) {
    int e = blockIdx.x * 256 + threadIdx.x;
    if (e >= E) return;
    int4 nb = nbr[e];
    const float* xs = x + (size_t)e * 11;
    const float* xa = x + (size_t)nb.x * 11;
    const float* xb = x + (size_t)nb.y * 11;
    const float* xc = x + (size_t)nb.z * 11;
    const float* xd = x + (size_t)nb.w * 11;
    float sf[11], af[11], bf[11], cf[11], df[11];
#pragma unroll
    for (int k = 0; k < 11; k++) { sf[k]=xs[k]; af[k]=xa[k]; bf[k]=xb[k]; cf[k]=xc[k]; df[k]=xd[k]; }
    float h[64];
#pragma unroll
    for (int n = 0; n < 64; n++) h[n] = b0[n];
#pragma unroll
    for (int k = 0; k < 11; k++) {
        float f0 = sf[k];
        float f1 = fabsf(af[k] - cf[k]);
        float f2 = af[k] + cf[k];
        float f3 = fabsf(bf[k] - df[k]);
        float f4 = bf[k] + df[k];
#pragma unroll
        for (int n = 0; n < 64; n++) {
            h[n] = fmaf(f0, w0[k * 64 + n],
                   fmaf(f1, w0[(11 + k) * 64 + n],
                   fmaf(f2, w0[(22 + k) * 64 + n],
                   fmaf(f3, w0[(33 + k) * 64 + n],
                   fmaf(f4, w0[(44 + k) * 64 + n], h[n])))));
        }
    }
    float s = 0.f, q = 0.f;
#pragma unroll
    for (int n = 0; n < 64; n++) { s += h[n]; q += h[n] * h[n]; }
    float mu   = s * (1.0f / 64.0f);
    float var  = q * (1.0f / 64.0f) - mu * mu;
    float rstd = rsqrtf(var + LN_EPS);
    float o[64];
#pragma unroll
    for (int n = 0; n < 64; n++) {
        float v = (h[n] - mu) * rstd * g[n] + be[n];
        o[n] = v > 0.f ? v : 0.f;
    }
    float4* xr4 = (float4*)(xres + (size_t)e * 64);
#pragma unroll
    for (int qd = 0; qd < 16; qd++)
        xr4[qd] = make_float4(o[4*qd], o[4*qd+1], o[4*qd+2], o[4*qd+3]);
    h8* xo8 = (h8*)(xout + (size_t)e * 64);
#pragma unroll
    for (int qd = 0; qd < 8; qd++) {
        h8 p;
#pragma unroll
        for (int j = 0; j < 8; j++) p[j] = (_Float16)o[8*qd + j];
        xo8[qd] = p;
    }
}

// ---------------------------------------------------------------------------
// Mesh-conv layers 1..3: gather f16 neighbors -> LDS feats (16x320/wave),
// GEMM via v_wmma_f32_16x16x32_f16 (10 K-steps x 4 N-tiles), LN+ReLU+residual.
// LAST fuses the 64->32->1 classifier and writes logits (skips writeback).
// ---------------------------------------------------------------------------
template <bool LAST>
__global__ __launch_bounds__(128)
void mesh_layer_kernel(const _Float16* __restrict__ xin, const int4* __restrict__ nbr,
                       const _Float16* __restrict__ wp, const float* __restrict__ bias,
                       const float* __restrict__ g, const float* __restrict__ be,
                       float* __restrict__ xres, _Float16* __restrict__ xout,
                       const float* __restrict__ cw1, const float* __restrict__ cb1,
                       const float* __restrict__ cw2, const float* __restrict__ cb2,
                       float* __restrict__ out, int E) {
    __shared__ __align__(16) _Float16 sF[4][16 * ROWH];
    const int wave = threadIdx.x >> 5;
    const int lane = threadIdx.x & 31;
    const int tileBase = blockIdx.x * 64 + wave * 16;
    const unsigned* xu = (const unsigned*)xin;   // 2 packed f16 per load
    _Float16* fbase = &sF[wave][0];

    // ---- build feats tile: 16 edges x 320 halfs (segments at 0,64,128,192,256)
    for (int i = 0; i < 16; i++) {
        int e = tileBase + i; if (e >= E) e = E - 1;
        int4 nb = nbr[e];
        unsigned us = xu[(size_t)e * 32 + lane];
        h2 a = __builtin_bit_cast(h2, xu[(size_t)nb.x * 32 + lane]);
        h2 b = __builtin_bit_cast(h2, xu[(size_t)nb.y * 32 + lane]);
        h2 c = __builtin_bit_cast(h2, xu[(size_t)nb.z * 32 + lane]);
        h2 d = __builtin_bit_cast(h2, xu[(size_t)nb.w * 32 + lane]);
        unsigned* row = (unsigned*)(fbase + i * ROWH);
        row[lane]       = us;
        row[32 + lane]  = __builtin_bit_cast(unsigned, h2abs(a - c));
        h2 s2 = a + c;
        row[64 + lane]  = __builtin_bit_cast(unsigned, s2);
        row[96 + lane]  = __builtin_bit_cast(unsigned, h2abs(b - d));
        h2 s4 = b + d;
        row[128 + lane] = __builtin_bit_cast(unsigned, s4);
    }

    // ---- WMMA: D(16x16) = A(16x32) x B(32x16), K=320, N=64
    const int nl = lane & 15;
    v8f acc[4];
#pragma unroll
    for (int t = 0; t < 4; t++) {
        float bv = bias[t * 16 + nl];
        v8f cv = {bv, bv, bv, bv, bv, bv, bv, bv};
        acc[t] = cv;
    }
    // A layout: lanes<16 hold row M=lane, K {kb..kb+7} and {kb+16..kb+23};
    //           lanes>=16 same row, K offsets +8.
    const _Float16* frow = fbase + nl * ROWH + ((lane >> 4) ? 8 : 0);
#pragma unroll
    for (int ks = 0; ks < 10; ks++) {
        h8 alo = *(const h8*)(frow + ks * 32);
        h8 ahi = *(const h8*)(frow + ks * 32 + 16);
        v16h aF = __builtin_shufflevector(alo, ahi, 0,1,2,3,4,5,6,7,8,9,10,11,12,13,14,15);
#pragma unroll
        for (int t = 0; t < 4; t++) {
            const h8* bq = (const h8*)(wp + ((ks * 4 + t) * 32 + lane) * 16);
            h8 blo = bq[0], bhi = bq[1];
            v16h bF = __builtin_shufflevector(blo, bhi, 0,1,2,3,4,5,6,7,8,9,10,11,12,13,14,15);
            acc[t] = __builtin_amdgcn_wmma_f32_16x16x32_f16(
                false, aF, false, bF, (short)0, acc[t], false, false);
        }
    }

    // ---- scatter D to LDS as [16 edges][64] f32 (reuse feats region; wave-local,
    //      LDS ops from the same wave are in-order)
    float* so = (float*)fbase;
    const int mbase = (lane >> 4) << 3;
#pragma unroll
    for (int t = 0; t < 4; t++) {
#pragma unroll
        for (int v = 0; v < 8; v++)
            so[(mbase + v) * 64 + t * 16 + nl] = acc[t][v];
    }

    // ---- LN + ReLU + residual: 2 lanes per edge (each owns 32 cols)
    const int el = lane >> 1;
    const int part = lane & 1;
    const float* r = so + el * 64 + part * 32;
    float s = 0.f, q = 0.f;
#pragma unroll
    for (int j = 0; j < 32; j++) { float v = r[j]; s += v; q += v * v; }
    s += __shfl_xor(s, 1, 32);
    q += __shfl_xor(q, 1, 32);
    float mu   = s * (1.0f / 64.0f);
    float var  = q * (1.0f / 64.0f) - mu * mu;
    float rstd = rsqrtf(var + LN_EPS);
    int e = tileBase + el;
    bool ok = e < E;
    size_t rowoff = (size_t)(ok ? e : 0) * 64 + part * 32;
    float nv[32];
#pragma unroll
    for (int j = 0; j < 32; j++) {
        int cidx = part * 32 + j;
        float v = (r[j] - mu) * rstd * g[cidx] + be[cidx];
        v = v > 0.f ? v : 0.f;
        nv[j] = v + xres[rowoff + j];       // residual: x = h + x
    }

    if (!LAST) {
        if (ok) {
#pragma unroll
            for (int j = 0; j < 32; j++) xres[rowoff + j] = nv[j];
            unsigned* xo = (unsigned*)xout + rowoff / 2;
#pragma unroll
            for (int j = 0; j < 16; j++) {
                h2 p = {(_Float16)nv[2 * j], (_Float16)nv[2 * j + 1]};
                xo[j] = __builtin_bit_cast(unsigned, p);
            }
        }
    } else {
        // stash final x back into LDS row, then fused classifier
        float* wr = so + el * 64 + part * 32;
#pragma unroll
        for (int j = 0; j < 32; j++) wr[j] = nv[j];
        const float* rr = so + el * 64;
        float lsum = 0.f;
#pragma unroll
        for (int jj = 0; jj < 16; jj++) {
            int jn = part * 16 + jj;
            float z = cb1[jn];
            for (int c2 = 0; c2 < 64; c2++)
                z = fmaf(rr[c2], cw1[c2 * 32 + jn], z);
            z = z > 0.f ? z : 0.f;
            lsum = fmaf(z, cw2[jn], lsum);
        }
        lsum += __shfl_xor(lsum, 1, 32);
        if (part == 0 && ok) out[e] = lsum + cb2[0];
    }
}

// ---------------------------------------------------------------------------
extern "C" void kernel_launch(void* const* d_in, const int* in_sizes, int n_in,
                              void* d_out, int out_size, void* d_ws, size_t ws_size,
                              hipStream_t stream) {
    const float* x      = (const float*)d_in[0];
    const int4*  nbr    = (const int4*)d_in[1];
    const float* w0     = (const float*)d_in[2];
    const float* b0     = (const float*)d_in[3];
    const float* w_rest = (const float*)d_in[4];
    const float* b_rest = (const float*)d_in[5];
    const float* ln_g   = (const float*)d_in[6];
    const float* ln_b   = (const float*)d_in[7];
    const float* cw1    = (const float*)d_in[8];
    const float* cb1    = (const float*)d_in[9];
    const float* cw2    = (const float*)d_in[10];
    const float* cb2    = (const float*)d_in[11];
    float* out = (float*)d_out;

    const int E = in_sizes[0] / 11;
    const size_t eh = (size_t)E * 64;
    char* ws = (char*)d_ws;
    float*    xres = (float*)ws;                      // E*64 f32 residual stream
    _Float16* xA   = (_Float16*)(ws + eh * 4);        // f16 gather buffer A
    _Float16* xB   = (_Float16*)(ws + eh * 6);        // f16 gather buffer B
    _Float16* wp   = (_Float16*)(ws + eh * 8);        // packed WMMA weights (3 layers)

    pack_w_kernel<<<240, 256, 0, stream>>>(w_rest, wp);
    layer0_kernel<<<(E + 255) / 256, 256, 0, stream>>>(x, nbr, w0, b0, ln_g, ln_b, xres, xA, E);

    int mb = (E + 63) / 64;
    mesh_layer_kernel<false><<<mb, 128, 0, stream>>>(xA, nbr, wp,         b_rest,       ln_g + 64,  ln_b + 64,
                                                     xres, xB, nullptr, nullptr, nullptr, nullptr, nullptr, E);
    mesh_layer_kernel<false><<<mb, 128, 0, stream>>>(xB, nbr, wp + 20480, b_rest + 64,  ln_g + 128, ln_b + 128,
                                                     xres, xA, nullptr, nullptr, nullptr, nullptr, nullptr, E);
    mesh_layer_kernel<true ><<<mb, 128, 0, stream>>>(xA, nbr, wp + 40960, b_rest + 128, ln_g + 192, ln_b + 192,
                                                     xres, nullptr, cw1, cb1, cw2, cb2, out, E);
}
